// StereoMatching_7842610282727
// MI455X (gfx1250) — matrix operand order repaired and checked by reading, not compiled, not run on hardware
//
#include <hip/hip_runtime.h>
#include <hip/hip_bf16.h>
#include <math.h>

#define HH 384
#define WW 640
#define DMAX 60
#define KKC 243.0f   // 9*9*3

typedef __attribute__((ext_vector_type(8)))  _Float16 v8h;
typedef __attribute__((ext_vector_type(16))) _Float16 v16h;
typedef __attribute__((ext_vector_type(8)))  float    v8f;

#define SQ_STRIDE 40   // f16 elements per halo row (80B: 16B-aligned, bank-spread)
#define SH_STRIDE 40   // f16 elements per transposed-hsum row

// ---------------------------------------------------------------------------
// 9x9 box sum of a 24x24 halo region -> 16x16 tile, via 3 WMMAs per tile.
//   pass 1 (x2): hsum[m][n] = sum_k q[m][k] * Band[k][n]  (Band[k][n]=1 iff n<=k<=n+8)
//   pass 2     : out [m][n] = sum_k Band[m][k] * hsum[k][n]
// One wave per tile. LDS is zero-padded so every operand read is unconditional
// (no exec-masked loads); EXEC stays all-ones for the WMMAs.
// ---------------------------------------------------------------------------
template <typename QF>
__device__ __forceinline__ v8f box9x9_tile(QF q, int lane) {
    __shared__ __align__(16) _Float16 sqh[32 * SQ_STRIDE]; // zero-padded halo (f16)
    __shared__ __align__(16) _Float16 sht[16 * SH_STRIDE]; // hsum, transposed [col][row]

    // zero the halo buffer: 32*40 = 1280 f16 = 640 dwords = 20 per lane
    {
        unsigned int* z = (unsigned int*)sqh;
#pragma unroll
        for (int i = 0; i < 20; ++i) z[i * 32 + lane] = 0u;
    }
    // fill 24x24 halo (576 = 18*32: every lane active every iteration)
#pragma unroll
    for (int i = 0; i < 18; ++i) {
        int idx = i * 32 + lane;
        int r = idx / 24, c = idx % 24;
        sqh[r * SQ_STRIDE + c] = (_Float16)q(r, c);
    }
    __syncthreads();

    const int nl   = lane & 15;
    const int c0   = (lane >= 16) ? 16 : 0;  // A-operand K base (contiguous layout)
    const int ro   = (lane >= 16) ? 8 : 0;   // C-layout row offset
    const int koff = (lane >= 16) ? 16 : 0;  // B-operand K base

    // horizontal band (B operand, 32x16): b[j] = 1 iff nl <= koff+j <= nl+8, k<24
    v16h bb;
#pragma unroll
    for (int j = 0; j < 16; ++j) {
        int k = koff + j;
        bb[j] = (_Float16)((k < 24 && k >= nl && k <= nl + 8) ? 1.0f : 0.0f);
    }

    v8f cz = {0.f, 0.f, 0.f, 0.f, 0.f, 0.f, 0.f, 0.f};

#pragma unroll
    for (int g = 0; g < 2; ++g) {
        int rr = 16 * g + nl;                       // region row held by this lane
        // A operand: two contiguous 16B LDS loads (K = c0..c0+7 and c0+8..c0+15)
        v8h a_lo = *(const v8h*)(sqh + rr * SQ_STRIDE + c0);
        v8h a_hi = *(const v8h*)(sqh + rr * SQ_STRIDE + c0 + 8);
        v16h aa = __builtin_shufflevector(a_lo, a_hi,
                    0, 1, 2, 3, 4, 5, 6, 7, 8, 9, 10, 11, 12, 13, 14, 15);
        v8f cc = __builtin_amdgcn_wmma_f32_16x16x32_f16(false, aa, false, bb,
                                                        (short)0, cz, false, false);
        // C layout: this lane holds 8 consecutive rows (16g+ro..+7) of column nl.
        // Rows >=24 are sums of zero padding -> naturally 0, store unconditionally.
        v8h hh;
#pragma unroll
        for (int v = 0; v < 8; ++v) hh[v] = (_Float16)cc[v];
        *(v8h*)(sht + nl * SH_STRIDE + 16 * g + ro) = hh;
    }
    __syncthreads();

    // vertical band as A operand: a[i] = 1 iff nl <= K <= nl+8 (K per A map, k<24)
    v16h ab;
#pragma unroll
    for (int i = 0; i < 16; ++i) {
        int k = i + c0;
        ab[i] = (_Float16)((k < 24 && k >= nl && k <= nl + 8) ? 1.0f : 0.0f);
    }
    // hsum column as B operand: 16 contiguous f16 = two 16B LDS loads
    v8h b_lo = *(const v8h*)(sht + nl * SH_STRIDE + koff);
    v8h b_hi = *(const v8h*)(sht + nl * SH_STRIDE + koff + 8);
    v16h bh = __builtin_shufflevector(b_lo, b_hi,
                0, 1, 2, 3, 4, 5, 6, 7, 8, 9, 10, 11, 12, 13, 14, 15);
    return __builtin_amdgcn_wmma_f32_16x16x32_f16(false, ab, false, bh,
                                                  (short)0, cz, false, false);
}

// ---------------------------------------------------------------------------
// Stage 1: per-image window stats  S = box9(sum_c img), S2 = box9(sum_c img^2)
// grid.z: 0=SL 1=S2L 2=SR 3=S2R ; 1 wave per 16x16 tile
// ---------------------------------------------------------------------------
__global__ void stereo_stats_kernel(const float* __restrict__ imgL,
                                    const float* __restrict__ imgR,
                                    float* __restrict__ ws) {
    int lane = threadIdx.x;
    int x0 = blockIdx.x * 16, y0 = blockIdx.y * 16;
    int z = blockIdx.z;
    const float* img = (z < 2) ? imgL : imgR;
    bool wantSq = (z & 1);

    auto q = [&](int r, int c) -> float {
        int y = y0 - 4 + r, x = x0 - 4 + c;
        if (y < 0 || y >= HH || x < 0 || x >= WW) return 0.0f;
        float a = img[(0 * HH + y) * WW + x];
        float b = img[(1 * HH + y) * WW + x];
        float d = img[(2 * HH + y) * WW + x];
        return wantSq ? (a * a + b * b + d * d) : (a + b + d);
    };

    v8f out = box9x9_tile(q, lane);

    float* dst = ws + (size_t)z * (HH * WW);
    int nl = lane & 15, ro = (lane >= 16) ? 8 : 0;
#pragma unroll
    for (int v = 0; v < 8; ++v) {
        int y = y0 + v + ro, x = x0 + nl;
        dst[y * WW + x] = out[v];
    }
}

// ---------------------------------------------------------------------------
// Stage 2: NCC cost volumes. grid.z = dir*60 + (d-1); 1 wave per 16x16 tile.
// ---------------------------------------------------------------------------
__global__ void stereo_cost_kernel(const float* __restrict__ imgL,
                                   const float* __restrict__ imgR,
                                   float* __restrict__ ws) {
    int lane = threadIdx.x;
    int x0 = blockIdx.x * 16, y0 = blockIdx.y * 16;
    int z = blockIdx.z;
    int dir = z / DMAX;                  // 0: costL (shift R by -d), 1: costR (+d)
    int d = (z % DMAX) + 1;
    const float* base = dir ? imgR : imgL;
    const float* oth  = dir ? imgL : imgR;
    int sgn = dir ? 1 : -1;

    auto q = [&](int r, int c) -> float {
        int y = y0 - 4 + r, x = x0 - 4 + c;
        if (y < 0 || y >= HH || x < 0 || x >= WW) return 0.0f;
        int xs = x + sgn * d;
        if (xs < 0 || xs >= WW) return 0.0f;
        float s = 0.0f;
#pragma unroll
        for (int ch = 0; ch < 3; ++ch)
            s += base[(ch * HH + y) * WW + x] * oth[(ch * HH + y) * WW + xs];
        return s;
    };

    v8f prod = box9x9_tile(q, lane);

    const size_t HW = (size_t)HH * WW;
    const float* bS  = ws + (size_t)(dir ? 2 : 0) * HW;
    const float* bS2 = ws + (size_t)(dir ? 3 : 1) * HW;
    const float* oS  = ws + (size_t)(dir ? 0 : 2) * HW;
    const float* oS2 = ws + (size_t)(dir ? 1 : 3) * HW;
    float* cost = ws + 4 * HW + (size_t)dir * DMAX * HW + (size_t)(d - 1) * HW;

    const float inv = 1.0f / KKC;
    int nl = lane & 15, ro = (lane >= 16) ? 8 : 0;
#pragma unroll
    for (int v = 0; v < 8; ++v) {
        int y = y0 + v + ro, x = x0 + nl;
        float baseS  = bS[y * WW + x];
        float baseS2 = bS2[y * WW + x];
        float baseA  = baseS * inv;
        float baseA2 = baseA * baseA;
        int xs = x + sgn * d;
        bool valid = (xs >= 0) && (xs < WW);
        int xc = valid ? xs : 0;
        float shS  = valid ? oS [y * WW + xc] : 0.0f;
        float shS2 = valid ? oS2[y * WW + xc] : 0.0f;
        float shA  = shS * inv;
        float shA2 = shA * shA;
        float prodSum = prod[v] - baseA * shS - shA * baseS + KKC * (baseA * shA);
        float n1 = sqrtf(baseS2 - 2.0f * baseA * baseS + KKC * baseA2 + 1e-5f);
        float n2 = sqrtf(shS2 - 2.0f * shA * shS + KKC * shA2 + 1e-5f);
        float corr = (prodSum + 1e-6f) / (n1 * n2 + 1e-6f);
        cost[y * WW + x] = valid ? corr : 0.0f;
    }
}

// ---------------------------------------------------------------------------
// Stage 3: softmax-weighted subpixel disparity (60 costs in registers)
// ---------------------------------------------------------------------------
__global__ void stereo_disp_kernel(const float* __restrict__ cost,
                                   float* __restrict__ disp,
                                   const float* __restrict__ beta_p) {
    int i = blockIdx.x * 256 + threadIdx.x;
    if (i >= HH * WW) return;
    float beta = beta_p[0];
    const size_t HW = (size_t)HH * WW;

    float c[DMAX];
    float m = -1e30f;
#pragma unroll
    for (int d = 0; d < DMAX; ++d) {
        c[d] = cost[(size_t)d * HW + i];
        m = fmaxf(m, c[d]);
    }
    float se = 0.0f, sv = 0.0f;
#pragma unroll
    for (int d = 0; d < DMAX; ++d) {
        float cp = (d > 0)        ? c[d - 1] : 0.0f;
        float cn = (d < DMAX - 1) ? c[d + 1] : 0.0f;
        float dv = (float)(d + 1) +
                   (cp - cn + 1e-6f) / (2.0f * (cp + cn - 2.0f * c[d]) + 1e-6f);
        float e = __expf((c[d] - m) * beta);
        se += e;
        sv += e * dv;
    }
    disp[i] = sv / se;
}

// ---------------------------------------------------------------------------
// Stage 4: 9x9 bilateral (normalizations cancel -> single fused exp weight)
// ---------------------------------------------------------------------------
__device__ __forceinline__ int reflect_idx(int i, int n) {
    if (i < 0) return -i;
    if (i >= n) return 2 * n - 2 - i;
    return i;
}

__global__ void stereo_bilateral_kernel(const float* __restrict__ in,
                                        float* __restrict__ out) {
    int i = blockIdx.x * 256 + threadIdx.x;
    if (i >= HH * WW) return;
    int y = i / WW, x = i % WW;
    float v0 = in[i];
    float num = 0.0f, den = 0.0f;
#pragma unroll
    for (int dy = -4; dy <= 4; ++dy) {
        int yy = reflect_idx(y + dy, HH);
#pragma unroll
        for (int dx = -4; dx <= 4; ++dx) {
            int xx = reflect_idx(x + dx, WW);
            float p = in[yy * WW + xx];
            float dc = p - v0;
            // 1/(2*0.05^2)=200 ; 1/(2*5^2)=0.02
            float w = __expf(-dc * dc * 200.0f - (float)(dy * dy + dx * dx) * 0.02f);
            num += w * p;
            den += w;
        }
    }
    out[i] = num / den;
}

// ---------------------------------------------------------------------------
// Stage 5: left-right consistency + depth
// ---------------------------------------------------------------------------
__global__ void stereo_lrc_depth_kernel(const float* __restrict__ dispL,
                                        const float* __restrict__ dispR,
                                        const float* __restrict__ f_p,
                                        const float* __restrict__ b_p,
                                        float* __restrict__ out) {
    int i = blockIdx.x * 256 + threadIdx.x;
    if (i >= HH * WW) return;
    int y = i / WW, x = i % WW;
    float dL = dispL[i];
    float xp = (float)x - dL;
    float x0f = floorf(xp);
    float wgt = xp - x0f;
    int x0 = (int)x0f;

    int xc0 = min(max(x0, 0), WW - 1);
    int xc1 = min(max(x0 + 1, 0), WW - 1);
    float m0 = (x0 >= 0 && x0 <= WW - 1) ? 1.0f : 0.0f;
    float m1 = (x0 + 1 >= 0 && x0 + 1 <= WW - 1) ? 1.0f : 0.0f;
    float v0 = dispR[y * WW + xc0] * m0;
    float v1 = dispR[y * WW + xc1] * m1;
    float warped = (1.0f - wgt) * v0 + wgt * v1;
    float maskv  = (1.0f - wgt) * m0 + wgt * m1;
    warped *= (maskv >= 0.9999f) ? 1.0f : 0.0f;
    float dd = dL - warped;
    float disp = (dd * dd > 0.5f) ? -1.0f : dL;
    out[i] = f_p[0] * b_p[0] / (disp + 1e-6f);
}

// ---------------------------------------------------------------------------
extern "C" void kernel_launch(void* const* d_in, const int* in_sizes, int n_in,
                              void* d_out, int out_size, void* d_ws, size_t ws_size,
                              hipStream_t stream) {
    (void)in_sizes; (void)n_in; (void)out_size; (void)ws_size;
    const float* imgL = (const float*)d_in[0];
    const float* imgR = (const float*)d_in[1];
    const float* f    = (const float*)d_in[2];
    const float* blD  = (const float*)d_in[3];
    const float* beta = (const float*)d_in[4];
    float* ws  = (float*)d_ws;
    float* out = (float*)d_out;

    const size_t HW = (size_t)HH * WW;
    // ws layout (floats): [0..4)HW stats SL,S2L,SR,S2R | [4..64)HW costL |
    // [64..124)HW costR | 124:dispL 125:dispR 126:dispLb 127:dispRb

    dim3 tiles(WW / 16, HH / 16, 4);
    stereo_stats_kernel<<<tiles, 32, 0, stream>>>(imgL, imgR, ws);

    dim3 ctiles(WW / 16, HH / 16, 2 * DMAX);
    stereo_cost_kernel<<<ctiles, 32, 0, stream>>>(imgL, imgR, ws);

    int nb = (HH * WW + 255) / 256;
    stereo_disp_kernel<<<nb, 256, 0, stream>>>(ws + 4 * HW,  ws + 124 * HW, beta);
    stereo_disp_kernel<<<nb, 256, 0, stream>>>(ws + 64 * HW, ws + 125 * HW, beta);

    stereo_bilateral_kernel<<<nb, 256, 0, stream>>>(ws + 124 * HW, ws + 126 * HW);
    stereo_bilateral_kernel<<<nb, 256, 0, stream>>>(ws + 125 * HW, ws + 127 * HW);

    stereo_lrc_depth_kernel<<<nb, 256, 0, stream>>>(ws + 126 * HW, ws + 127 * HW,
                                                    f, blD, out);
}
